// XGNN_Equi_force_ckpt_23313082483595
// MI455X (gfx1250) — compile-verified, matrix operand order checked by reading, not run on hardware
//
#include <hip/hip_runtime.h>
#include <hip/hip_bf16.h>
#include <math.h>

// ---------------------------------------------------------------------------
// Types for CDNA5 WMMA (wave32): D(f32 16x16) = A(bf16 16x32) x B(bf16 32x16) + C
// ---------------------------------------------------------------------------
typedef __attribute__((ext_vector_type(16))) __bf16 v16bf;
typedef __attribute__((ext_vector_type(8)))  float  v8f;

#define HDIM 128
#define FDIM 169
#define FPAD 192      // K-pad (multiple of 32) for edge_attr features
#define SHDIM 352     // 352 = 32*11 (already K-aligned)
#define RBFD 20
#define RBFP 32
#define NGRAPH 64
#define NLAYER 3
#define CUT 5.0f
#define PI_F 3.14159265358979323846f

__device__ __forceinline__ unsigned short f2bf(float f) {
    unsigned int u = __float_as_uint(f);
    unsigned int r = (u + 0x7FFFu + ((u >> 16) & 1u)) >> 16;   // RNE
    return (unsigned short)r;
}
__device__ __forceinline__ float sigm(float x) { return 1.0f / (1.0f + __expf(-x)); }
__device__ __forceinline__ float silu(float x) { return x * sigm(x); }
__device__ __forceinline__ float dsilu(float x) {
    float s = sigm(x);
    return s * (1.0f + x * (1.0f - s));
}

struct Frag { uint4 lo, hi; };   // 32B = one bf16 16x32 fragment per lane

// ---------------------------------------------------------------------------
// WMMA GEMM: C[M x N] = A[M x Kp](bf16, lda=Kp) * B (bf16, stored TRANSPOSED:
// Bt[n][k], n padded to gridDim.y*128 rows, row stride Kp) (+bias) (+=C if beta)
//
// Tile 128x128, K step 32, 256 threads = 8 waves; wave w owns rows [16w,16w+16).
// Double-buffered async global->LDS pipeline (ASYNCcnt): stage s+1's copies run
// while stage s is consumed by WMMAs. Each wave/thread issues exactly 4 async
// b128 loads per stage, so "s_wait_asynccnt 4" == previous stage landed.
// B fragments are software-pipelined through 3 rotating registers so ds_load
// latency overlaps the WMMA chain (waits become dscnt<=4, not 0).
// LDS per buffer: As = [0,10240) 128 rows x 80B; Bs = [10240,20480) same.
// Requires gridDim.x == M/128 (M multiple of 128) -> no staging guards.
// ---------------------------------------------------------------------------
#define GEMM_LDS_BYTES 40960

__global__ __launch_bounds__(256) void k_gemm(
    const unsigned short* __restrict__ A, int lda,
    const unsigned short* __restrict__ Bt,
    float* __restrict__ C, int ldc,
    int M, int N, int Kp,
    const float* __restrict__ bias, int beta)
{
    extern __shared__ unsigned short lds[];

    const int m0   = blockIdx.x * 128;
    const int n0   = blockIdx.y * 128;
    const int t    = threadIdx.x;
    const int w    = t >> 5;
    const int lane = t & 31;
    const int mrow = lane & 15;
    const int half = lane >> 4;

    // staging coordinates: 2 threads per 128-row tile row, 32B each
    const int srow  = t >> 1;
    const int shalf = t & 1;
    const unsigned a_lds = (unsigned)srow * 80u + (unsigned)shalf * 32u;           // in As
    const unsigned b_lds = 10240u + (unsigned)srow * 80u + (unsigned)shalf * 32u;  // in Bs
    const unsigned short* ap = A  + (size_t)(m0 + srow) * lda + shalf * 16;
    const unsigned short* bp = Bt + (size_t)(n0 + srow) * Kp  + shalf * 16;

    v8f acc[8];
#pragma unroll
    for (int i = 0; i < 8; i++)
        for (int j = 0; j < 8; j++) acc[i][j] = 0.0f;

    const int nsteps = Kp >> 5;

    // async-issue one 32-K stage into LDS buffer at byte offset `buf`
    auto issue = [&](unsigned buf) {
        unsigned al = a_lds + buf, bl = b_lds + buf;
        asm volatile("global_load_async_to_lds_b128 %0, %1, off"
                     :: "v"(al), "v"(ap) : "memory");
        asm volatile("global_load_async_to_lds_b128 %0, %1, off offset:16"
                     :: "v"(al), "v"(ap) : "memory");
        asm volatile("global_load_async_to_lds_b128 %0, %1, off"
                     :: "v"(bl), "v"(bp) : "memory");
        asm volatile("global_load_async_to_lds_b128 %0, %1, off offset:16"
                     :: "v"(bl), "v"(bp) : "memory");
        ap += 32; bp += 32;
    };
    // load one B fragment (column nt*16+mrow, two 8-K runs) from buffer `base`
    auto ldB = [&](unsigned base, int nt) -> Frag {
        Frag f;
        unsigned bb = base + 10240u + (unsigned)(nt * 16 + mrow) * 80u + (unsigned)half * 16u;
        f.lo = *(const uint4*)((const char*)lds + bb);
        f.hi = *(const uint4*)((const char*)lds + bb + 32u);
        return f;
    };
    auto mm = [&](v16bf af, const Frag& f, v8f c) -> v8f {
        return __builtin_amdgcn_wmma_f32_16x16x32_bf16(
            false, af, false, __builtin_bit_cast(v16bf, f), (short)0, c, false, false);
    };
    // consume stage resident in buffer `base` (A frag + 8 pipelined B frags)
    auto compute = [&](unsigned base) {
        Frag ar;
        unsigned ab = base + (unsigned)(w * 16 + mrow) * 80u + (unsigned)half * 16u;
        ar.lo = *(const uint4*)((const char*)lds + ab);
        ar.hi = *(const uint4*)((const char*)lds + ab + 32u);
        v16bf af = __builtin_bit_cast(v16bf, ar);
        Frag f0 = ldB(base, 0);
        Frag f1 = ldB(base, 1);
        Frag f2 = ldB(base, 2); acc[0] = mm(af, f0, acc[0]);
        f0 = ldB(base, 3);      acc[1] = mm(af, f1, acc[1]);
        f1 = ldB(base, 4);      acc[2] = mm(af, f2, acc[2]);
        f2 = ldB(base, 5);      acc[3] = mm(af, f0, acc[3]);
        f0 = ldB(base, 6);      acc[4] = mm(af, f1, acc[4]);
        f1 = ldB(base, 7);      acc[5] = mm(af, f2, acc[5]);
        acc[6] = mm(af, f0, acc[6]);
        acc[7] = mm(af, f1, acc[7]);
    };

    issue(0u);                                   // stage 0 -> buffer 0
    unsigned cur = 0u;                           // buffer holding current stage
    for (int s = 0; s < nsteps - 1; s++) {       // steady state
        issue(cur ^ 20480u);                     // stage s+1 -> other buffer
        asm volatile("s_wait_asynccnt 0x4" ::: "memory");  // stage s landed
        __syncthreads();
        compute(cur);
        __syncthreads();                         // readers done before overwrite
        cur ^= 20480u;
    }
    asm volatile("s_wait_asynccnt 0x0" ::: "memory");      // last stage landed
    __syncthreads();
    compute(cur);

    // ---- epilogue (C/D layout: lane half selects M+8, VGPR r = M row)
#pragma unroll
    for (int nt = 0; nt < 8; nt++) {
        int col = n0 + nt * 16 + mrow;
        if (col >= N) continue;
        float bb = bias ? bias[col] : 0.0f;
#pragma unroll
        for (int r = 0; r < 8; r++) {
            int row = m0 + w * 16 + half * 8 + r;
            if (row >= M) continue;
            float val = acc[nt][r] + bb;
            size_t off = (size_t)row * ldc + col;
            if (beta) val += C[off];
            C[off] = val;
        }
    }
}

// ---------------------------------------------------------------------------
// Elementwise / utility kernels
// ---------------------------------------------------------------------------
__global__ void k_zero(float* p, long long n) {
    long long i = (long long)blockIdx.x * blockDim.x + threadIdx.x;
    if (i < n) p[i] = 0.0f;
}
__global__ void k_f32_to_bf16(const float* __restrict__ x, unsigned short* __restrict__ o, long long n) {
    long long i = (long long)blockIdx.x * blockDim.x + threadIdx.x;
    if (i < n) o[i] = f2bf(x[i]);
}
__global__ void k_silu_bf16(const float* __restrict__ x, unsigned short* __restrict__ o, long long n) {
    long long i = (long long)blockIdx.x * blockDim.x + threadIdx.x;
    if (i < n) o[i] = f2bf(silu(x[i]));
}
__global__ void k_silu_f32(const float* __restrict__ x, float* __restrict__ o, long long n) {
    long long i = (long long)blockIdx.x * blockDim.x + threadIdx.x;
    if (i < n) o[i] = silu(x[i]);
}
__global__ void k_add_silu(const float* __restrict__ base, const float* __restrict__ pre,
                           float* __restrict__ o, long long n) {
    long long i = (long long)blockIdx.x * blockDim.x + threadIdx.x;
    if (i < n) o[i] = base[i] + silu(pre[i]);
}
__global__ void k_dsilu_bf16(const float* __restrict__ g, const float* __restrict__ pre,
                             unsigned short* __restrict__ o, long long n) {
    long long i = (long long)blockIdx.x * blockDim.x + threadIdx.x;
    if (i < n) o[i] = f2bf(g[i] * dsilu(pre[i]));
}
// weight fp32 -> bf16, emitted TRANSPOSED for the GEMM B operand:
//   out[r][c] over NpG x KpG (r = gemm-N index, c = gemm-K index)
//   trans=0: B = W      (Kw x Nw): out[r][c] = W[c][r]
//   trans=1: B = W^T              : out[r][c] = W[r][c]
__global__ void k_wconv(const float* __restrict__ W, unsigned short* __restrict__ o,
                        int Kw, int Nw, int KpG, int NpG, int trans) {
    long long i = (long long)blockIdx.x * blockDim.x + threadIdx.x;
    if (i >= (long long)KpG * NpG) return;
    int r = (int)(i / KpG), c = (int)(i % KpG);
    float v = 0.0f;
    if (trans) { if (r < Kw && c < Nw) v = W[(size_t)r * Nw + c]; }
    else       { if (c < Kw && r < Nw) v = W[(size_t)c * Nw + r]; }
    o[i] = f2bf(v);
}
// Ea = (edge_attr * env) -> bf16, padded to FPAD
__global__ void k_ea(const float* __restrict__ eattr, const float* __restrict__ env,
                     unsigned short* __restrict__ o, int E) {
    long long i = (long long)blockIdx.x * blockDim.x + threadIdx.x;
    if (i >= (long long)E * FPAD) return;
    int e = (int)(i / FPAD), k = (int)(i % FPAD);
    float v = (k < FDIM) ? eattr[(size_t)e * FDIM + k] * env[e] : 0.0f;
    o[i] = f2bf(v);
}
// g_x init: grad of sum(e_edge) wrt x_final = w_out[c] * env[e]
__global__ void k_gxinit(const float* __restrict__ env, const float* __restrict__ wout,
                         float* __restrict__ gx, int E) {
    long long i = (long long)blockIdx.x * blockDim.x + threadIdx.x;
    if (i >= (long long)E * HDIM) return;
    int e = (int)(i >> 7), c = (int)(i & 127);
    gx[i] = wout[c] * env[e];
}

// ---------------------------------------------------------------------------
// Edge geometry: bond vec, d, envelope(+deriv), bessel rbf (f32 + bf16),
// spherical harmonics (bf16 for GEMM)
// ---------------------------------------------------------------------------
__global__ void k_geom(const float* __restrict__ pos, const int* __restrict__ eidx,
                       float* ux, float* uy, float* uz, float* dd,
                       float* env, float* denv,
                       float* rbf, unsigned short* rbfb, unsigned short* rshb, int E) {
    int e = blockIdx.x * blockDim.x + threadIdx.x;
    if (e >= E) return;
    int s = eidx[e], dt = eidx[E + e];
    float bx = pos[s * 3 + 0] - pos[dt * 3 + 0];
    float by = pos[s * 3 + 1] - pos[dt * 3 + 1];
    float bz = pos[s * 3 + 2] - pos[dt * 3 + 2];
    float d  = sqrtf(bx * bx + by * by + bz * bz + 1e-12f);
    float inv = 1.0f / d;
    float x = bx * inv, y = by * inv, z = bz * inv;
    ux[e] = x; uy[e] = y; uz[e] = z; dd[e] = d;

    float tt = d * (1.0f / CUT);
    float t2 = tt * tt, t4 = t2 * t2, t5 = t4 * tt, t6 = t5 * tt, t7 = t6 * tt;
    float ev = 0.0f, dev = 0.0f;
    if (d < CUT) {
        ev  = 1.0f - 21.0f * t5 + 35.0f * t6 - 15.0f * t7;
        dev = (-105.0f * t4 + 210.0f * t5 - 105.0f * t6) * (1.0f / CUT);
    }
    env[e] = ev; denv[e] = dev;

    float ds = fmaxf(d, 1e-6f);
    const float c0 = 0.6324555320336759f;              // sqrt(2/CUT)
    for (int n = 1; n <= RBFD; n++) {
        float a = (float)n * (PI_F / CUT);
        float v = c0 * __sinf(a * ds) / ds;
        rbf[(size_t)e * RBFD + n - 1] = v;
        rbfb[(size_t)e * RBFP + n - 1] = f2bf(v);
    }
    for (int n = RBFD; n < RBFP; n++) rbfb[(size_t)e * RBFP + n] = 0;

    unsigned short hy = f2bf(y), hz = f2bf(z), hx = f2bf(x);
    unsigned short* rb = rshb + (size_t)e * SHDIM;
    for (int r = 0; r < 64; r++) { rb[3 * r] = hy; rb[3 * r + 1] = hz; rb[3 * r + 2] = hx; }
    const float s3 = 1.7320508075688772f;
    unsigned short h2[5];
    h2[0] = f2bf(s3 * x * y);
    h2[1] = f2bf(s3 * y * z);
    h2[2] = f2bf(0.5f * (3.0f * z * z - 1.0f));
    h2[3] = f2bf(s3 * x * z);
    h2[4] = f2bf(0.5f * s3 * (x * x - y * y));
    for (int r = 0; r < 32; r++)
        for (int j = 0; j < 5; j++) rb[192 + 5 * r + j] = h2[j];
}

// ---------------------------------------------------------------------------
// Message passing: one wave per line-edge, 32 lanes x float4 = 128 cols
// ---------------------------------------------------------------------------
__global__ void k_msg_fwd(const float* __restrict__ x, const float* __restrict__ gate,
                          const int* __restrict__ le, float* __restrict__ agg, int LE) {
    long long i = (long long)blockIdx.x * blockDim.x + threadIdx.x;
    int e = (int)(i >> 5), lane = (int)(i & 31);
    if (e >= LE) return;
    int s = le[e], dt = le[LE + e];
    float4 xv = ((const float4*)(x    + (size_t)s * HDIM))[lane];
    float4 gv = ((const float4*)(gate + (size_t)s * HDIM))[lane];
    float* a = agg + (size_t)dt * HDIM + lane * 4;
    atomicAdd(a + 0, xv.x * gv.x);
    atomicAdd(a + 1, xv.y * gv.y);
    atomicAdd(a + 2, xv.z * gv.z);
    atomicAdd(a + 3, xv.w * gv.w);
}
__global__ void k_msg_bwd(const float* __restrict__ gagg, const float* __restrict__ gate,
                          const float* __restrict__ x, const int* __restrict__ le,
                          float* __restrict__ gx, float* __restrict__ gg, int LE) {
    long long i = (long long)blockIdx.x * blockDim.x + threadIdx.x;
    int e = (int)(i >> 5), lane = (int)(i & 31);
    if (e >= LE) return;
    int s = le[e], dt = le[LE + e];
    float4 gm = ((const float4*)(gagg + (size_t)dt * HDIM))[lane];
    float4 gv = ((const float4*)(gate + (size_t)s * HDIM))[lane];
    float4 xv = ((const float4*)(x    + (size_t)s * HDIM))[lane];
    float* px = gx + (size_t)s * HDIM + lane * 4;
    float* pg = gg + (size_t)s * HDIM + lane * 4;
    atomicAdd(px + 0, gm.x * gv.x); atomicAdd(px + 1, gm.y * gv.y);
    atomicAdd(px + 2, gm.z * gv.z); atomicAdd(px + 3, gm.w * gv.w);
    atomicAdd(pg + 0, gm.x * xv.x); atomicAdd(pg + 1, gm.y * xv.y);
    atomicAdd(pg + 2, gm.z * xv.z); atomicAdd(pg + 3, gm.w * xv.w);
}

// ---------------------------------------------------------------------------
// Output head: e_edge = (x . w_out) * env ; results[batch[src]] += e_edge
// also save dot for backward env-grad
// ---------------------------------------------------------------------------
__global__ void k_edge_out(const float* __restrict__ x3, const float* __restrict__ wout,
                           const float* __restrict__ env, const int* __restrict__ eidx,
                           const int* __restrict__ abatch, float* __restrict__ results,
                           float* __restrict__ envg, int E) {
    int e = blockIdx.x * blockDim.x + threadIdx.x;
    if (e >= E) return;
    const float* xr = x3 + (size_t)e * HDIM;
    float s = 0.0f;
    for (int c = 0; c < HDIM; c++) s += xr[c] * wout[c];
    envg[e] = s;
    atomicAdd(&results[abatch[eidx[e]]], s * env[e]);
}

// ---------------------------------------------------------------------------
// Per-edge backward of geometry: combine g_env, g_rbf, g_rsh -> g_bond,
// scatter into force accumulator. Also converts raw grad -> e_a_g (in-place).
// ---------------------------------------------------------------------------
__global__ void k_edge_bwd(const float* __restrict__ eattr, float* __restrict__ gea,
                           const float* __restrict__ grsh, const float* __restrict__ grbf,
                           const float* __restrict__ envg,
                           const float* ux, const float* uy, const float* uz,
                           const float* dd, const float* env, const float* denv,
                           const int* __restrict__ eidx, float* __restrict__ facc, int E) {
    int e = blockIdx.x * blockDim.x + threadIdx.x;
    if (e >= E) return;
    float envv = env[e];
    float g_env = envg[e];                     // from e_edge = dot * env
    float* gw = gea + (size_t)e * FDIM;
    const float* ea = eattr + (size_t)e * FDIM;
    for (int f = 0; f < FDIM; f++) {
        float g = gw[f];                       // grad wrt (eattr*env)
        g_env += g * ea[f];
        gw[f] = g * envv;                      // e_a_g (grad wrt edge_attr)
    }
    float d = dd[e];
    float gd = g_env * denv[e];
    float ds = fmaxf(d, 1e-6f);
    if (d > 1e-6f) {
        const float c0 = 0.6324555320336759f;
        for (int n = 1; n <= RBFD; n++) {
            float a = (float)n * (PI_F / CUT);
            float sn = __sinf(a * ds), cn = __cosf(a * ds);
            gd += grbf[(size_t)e * RBFD + n - 1] * c0 * (a * cn / ds - sn / (ds * ds));
        }
    }
    float x = ux[e], y = uy[e], z = uz[e];
    const float* gs = grsh + (size_t)e * SHDIM;
    float gxv = 0, gy = 0, gz = 0;
    for (int r = 0; r < 64; r++) { gy += gs[3 * r]; gz += gs[3 * r + 1]; gxv += gs[3 * r + 2]; }
    float g2[5] = {0, 0, 0, 0, 0};
    for (int r = 0; r < 32; r++) {
        const float* p = gs + 192 + 5 * r;
        g2[0] += p[0]; g2[1] += p[1]; g2[2] += p[2]; g2[3] += p[3]; g2[4] += p[4];
    }
    const float s3 = 1.7320508075688772f;
    gxv += s3 * y * g2[0] + s3 * z * g2[3] + s3 * x * g2[4];
    gy  += s3 * x * g2[0] + s3 * z * g2[1] - s3 * y * g2[4];
    gz  += s3 * y * g2[1] + 3.0f * z * g2[2] + s3 * x * g2[3];
    float udg = x * gxv + y * gy + z * gz;
    float invd = 1.0f / d;
    float gb0 = (gxv - x * udg) * invd + gd * x;
    float gb1 = (gy  - y * udg) * invd + gd * y;
    float gb2 = (gz  - z * udg) * invd + gd * z;
    int s = eidx[e], dt = eidx[E + e];
    atomicAdd(&facc[s * 3 + 0], gb0);  atomicAdd(&facc[s * 3 + 1], gb1);  atomicAdd(&facc[s * 3 + 2], gb2);
    atomicAdd(&facc[dt * 3 + 0], -gb0); atomicAdd(&facc[dt * 3 + 1], -gb1); atomicAdd(&facc[dt * 3 + 2], -gb2);
}

// edge_grads[e,k] = sum_f edge_attr_grad[e,k,f] * e_a_g[e,f];  S scatter
__global__ void k_einsum(const float* __restrict__ eag, const float* __restrict__ gea,
                         const int* __restrict__ eidx, float* __restrict__ facc, int E) {
    long long i = (long long)blockIdx.x * blockDim.x + threadIdx.x;
    if (i >= (long long)E * 3) return;
    int e = (int)(i / 3), k = (int)(i % 3);
    const float* gr = eag + (size_t)e * 3 * FDIM + (size_t)k * FDIM;
    const float* gw = gea + (size_t)e * FDIM;
    float s = 0.0f;
    for (int f = 0; f < FDIM; f++) s += gr[f] * gw[f];
    atomicAdd(&facc[eidx[e] * 3 + k], s);
    atomicAdd(&facc[eidx[E + e] * 3 + k], -s);
}
__global__ void k_force_out(const float* __restrict__ facc, float* __restrict__ out, long long n) {
    long long i = (long long)blockIdx.x * blockDim.x + threadIdx.x;
    if (i < n) out[i] = -facc[i];
}

// ---------------------------------------------------------------------------
// Host orchestration
// ---------------------------------------------------------------------------
static inline int cdiv(long long a, long long b) { return (int)((a + b - 1) / b); }

static void gemm(const unsigned short* A, int lda, const unsigned short* Bt,
                 float* C, int ldc, int M, int N, int Kp, const float* bias, int beta,
                 hipStream_t s) {
    dim3 g((unsigned)(M / 128), (unsigned)((N + 127) / 128));
    k_gemm<<<g, dim3(256), GEMM_LDS_BYTES, s>>>(A, lda, Bt, C, ldc, M, N, Kp, bias, beta);
}

extern "C" void kernel_launch(void* const* d_in, const int* in_sizes, int n_in,
                              void* d_out, int out_size, void* d_ws, size_t ws_size,
                              hipStream_t stream) {
    const float* atom_pos  = (const float*)d_in[0];
    const float* edge_attr = (const float*)d_in[1];
    const float* eag       = (const float*)d_in[2];   // edge_attr_grad (E,3,169)
    const float* W_mat     = (const float*)d_in[3];
    const float* b_mat     = (const float*)d_in[4];
    const float* W_emb     = (const float*)d_in[5];
    const float* b_emb     = (const float*)d_in[6];
    const float* Wv        = (const float*)d_in[7];   // (352,128)
    const float* Wr        = (const float*)d_in[8];   // (3,20,128)
    const float* Wx        = (const float*)d_in[9];   // (3,128,128)
    const float* w_out     = (const float*)d_in[10];  // (128,)
    const int*   eidx      = (const int*)d_in[11];    // (2,E)
    const int*   nei       = (const int*)d_in[12];    // (2,LE)
    const int*   abatch    = (const int*)d_in[13];    // (N,)

    const int E  = in_sizes[1] / FDIM;
    const int NN = in_sizes[0] / 3;
    const int LE = in_sizes[12] / 2;
    float* out_f = (float*)d_out;

    // ---- workspace carve-out --------------------------------------------
    char* wsp = (char*)d_ws;
    auto alloc = [&](size_t bytes) -> void* {
        void* p = (void*)wsp;
        wsp += (bytes + 255) & ~(size_t)255;
        return p;
    };
    float* g_ux   = (float*)alloc((size_t)E * 4);
    float* g_uy   = (float*)alloc((size_t)E * 4);
    float* g_uz   = (float*)alloc((size_t)E * 4);
    float* g_dd   = (float*)alloc((size_t)E * 4);
    float* g_env  = (float*)alloc((size_t)E * 4);
    float* g_denv = (float*)alloc((size_t)E * 4);
    float* g_envg = (float*)alloc((size_t)E * 4);
    float* g_rbf  = (float*)alloc((size_t)E * RBFD * 4);
    float* g_grbf = (float*)alloc((size_t)E * RBFD * 4);
    unsigned short* b_rbf = (unsigned short*)alloc((size_t)E * RBFP * 2);
    unsigned short* b_rsh = (unsigned short*)alloc((size_t)E * SHDIM * 2);
    unsigned short* b_ea  = (unsigned short*)alloc((size_t)E * FPAD * 2);
    unsigned short* b_x   = (unsigned short*)alloc((size_t)E * HDIM * 2);  // generic bf16 GEMM input
    float* P1   = (float*)alloc((size_t)E * HDIM * 4);
    float* SV   = (float*)alloc((size_t)E * HDIM * 4);
    float* GP   = (float*)alloc((size_t)E * HDIM * 4);
    float* GATE = (float*)alloc((size_t)E * HDIM * 4);
    float* AGG  = (float*)alloc((size_t)E * HDIM * 4);
    float* GX   = (float*)alloc((size_t)E * HDIM * 4);
    float* GG   = (float*)alloc((size_t)E * HDIM * 4);
    float* X[NLAYER + 1];
    for (int l = 0; l <= NLAYER; l++) X[l] = (float*)alloc((size_t)E * HDIM * 4);
    float* Y[NLAYER];
    for (int l = 0; l < NLAYER; l++) Y[l] = (float*)alloc((size_t)E * HDIM * 4);
    float* GRSH = (float*)alloc((size_t)E * SHDIM * 4);
    float* GEA  = (float*)alloc((size_t)E * FDIM * 4);
    float* FACC = (float*)alloc((size_t)NN * 3 * 4);
    // B operands, stored transposed Bt[n][k], n padded to ntiles*128
    unsigned short* bW_mat  = (unsigned short*)alloc((size_t)128 * FPAD * 2);   // N=128,K=192
    unsigned short* bW_matT = (unsigned short*)alloc((size_t)256 * HDIM * 2);   // N=169->256,K=128
    unsigned short* bW_emb  = (unsigned short*)alloc((size_t)128 * HDIM * 2);
    unsigned short* bW_embT = (unsigned short*)alloc((size_t)128 * HDIM * 2);
    unsigned short* bWv     = (unsigned short*)alloc((size_t)128 * SHDIM * 2);  // N=128,K=352
    unsigned short* bWvT    = (unsigned short*)alloc((size_t)384 * HDIM * 2);   // N=352->384,K=128
    unsigned short* bWr     = (unsigned short*)alloc((size_t)NLAYER * 128 * RBFP * 2); // N=128,K=32
    unsigned short* bWrT    = (unsigned short*)alloc((size_t)NLAYER * 128 * HDIM * 2); // N=20->128,K=128
    unsigned short* bWx     = (unsigned short*)alloc((size_t)NLAYER * 128 * HDIM * 2);
    unsigned short* bWxT    = (unsigned short*)alloc((size_t)NLAYER * 128 * HDIM * 2);

    const int B256 = 256;
    const long long EH = (long long)E * HDIM;

    // ---- weight conversion (bf16, transposed layout, padded) ------------
    k_wconv<<<cdiv((long long)128 * FPAD, B256), B256, 0, stream>>>(W_mat, bW_mat, FDIM, HDIM, FPAD, 128, 0);
    k_wconv<<<cdiv((long long)256 * HDIM, B256), B256, 0, stream>>>(W_mat, bW_matT, FDIM, HDIM, HDIM, 256, 1);
    k_wconv<<<cdiv((long long)128 * HDIM, B256), B256, 0, stream>>>(W_emb, bW_emb, HDIM, HDIM, HDIM, 128, 0);
    k_wconv<<<cdiv((long long)128 * HDIM, B256), B256, 0, stream>>>(W_emb, bW_embT, HDIM, HDIM, HDIM, 128, 1);
    k_wconv<<<cdiv((long long)128 * SHDIM, B256), B256, 0, stream>>>(Wv, bWv, SHDIM, HDIM, SHDIM, 128, 0);
    k_wconv<<<cdiv((long long)384 * HDIM, B256), B256, 0, stream>>>(Wv, bWvT, SHDIM, HDIM, HDIM, 384, 1);
    for (int l = 0; l < NLAYER; l++) {
        k_wconv<<<cdiv((long long)128 * RBFP, B256), B256, 0, stream>>>(
            Wr + (size_t)l * RBFD * HDIM, bWr + (size_t)l * 128 * RBFP, RBFD, HDIM, RBFP, 128, 0);
        k_wconv<<<cdiv((long long)128 * HDIM, B256), B256, 0, stream>>>(
            Wr + (size_t)l * RBFD * HDIM, bWrT + (size_t)l * 128 * HDIM, RBFD, HDIM, HDIM, 128, 1);
        k_wconv<<<cdiv((long long)128 * HDIM, B256), B256, 0, stream>>>(
            Wx + (size_t)l * HDIM * HDIM, bWx + (size_t)l * 128 * HDIM, HDIM, HDIM, HDIM, 128, 0);
        k_wconv<<<cdiv((long long)128 * HDIM, B256), B256, 0, stream>>>(
            Wx + (size_t)l * HDIM * HDIM, bWxT + (size_t)l * 128 * HDIM, HDIM, HDIM, HDIM, 128, 1);
    }

    // ---- forward --------------------------------------------------------
    k_geom<<<cdiv(E, B256), B256, 0, stream>>>(atom_pos, eidx, g_ux, g_uy, g_uz, g_dd,
                                               g_env, g_denv, g_rbf, b_rbf, b_rsh, E);
    k_ea<<<cdiv((long long)E * FPAD, B256), B256, 0, stream>>>(edge_attr, g_env, b_ea, E);

    gemm(b_ea, FPAD, bW_mat, P1, HDIM, E, HDIM, FPAD, b_mat, 0, stream);             // P1
    k_silu_bf16<<<cdiv(EH, B256), B256, 0, stream>>>(P1, b_x, EH);
    gemm(b_x, HDIM, bW_emb, X[0], HDIM, E, HDIM, HDIM, b_emb, 0, stream);            // neo_x
    gemm(b_rsh, SHDIM, bWv, SV, HDIM, E, HDIM, SHDIM, nullptr, 0, stream);           // rsh@Wv
    k_add_silu<<<cdiv(EH, B256), B256, 0, stream>>>(X[0], SV, X[0], EH);             // x0

    for (int l = 0; l < NLAYER; l++) {
        gemm(b_rbf, RBFP, bWr + (size_t)l * 128 * RBFP, GP, HDIM, E, HDIM, RBFP, nullptr, 0, stream);
        k_silu_f32<<<cdiv(EH, B256), B256, 0, stream>>>(GP, GATE, EH);
        k_zero<<<cdiv(EH, B256), B256, 0, stream>>>(AGG, EH);
        k_msg_fwd<<<cdiv((long long)LE * 32, B256), B256, 0, stream>>>(X[l], GATE, nei, AGG, LE);
        k_f32_to_bf16<<<cdiv(EH, B256), B256, 0, stream>>>(AGG, b_x, EH);
        gemm(b_x, HDIM, bWx + (size_t)l * 128 * HDIM, Y[l], HDIM, E, HDIM, HDIM, nullptr, 0, stream);
        k_add_silu<<<cdiv(EH, B256), B256, 0, stream>>>(X[l], Y[l], X[l + 1], EH);
    }

    k_zero<<<1, B256, 0, stream>>>(out_f, NGRAPH);
    k_edge_out<<<cdiv(E, B256), B256, 0, stream>>>(X[NLAYER], w_out, g_env, eidx, abatch,
                                                   out_f, g_envg, E);

    // ---- backward -------------------------------------------------------
    k_gxinit<<<cdiv(EH, B256), B256, 0, stream>>>(g_env, w_out, GX, E);
    k_zero<<<cdiv((long long)E * RBFD, B256), B256, 0, stream>>>(g_grbf, (long long)E * RBFD);

    for (int l = NLAYER - 1; l >= 0; l--) {
        k_dsilu_bf16<<<cdiv(EH, B256), B256, 0, stream>>>(GX, Y[l], b_x, EH);
        gemm(b_x, HDIM, bWxT + (size_t)l * 128 * HDIM, AGG, HDIM, E, HDIM, HDIM, nullptr, 0, stream); // g_agg
        gemm(b_rbf, RBFP, bWr + (size_t)l * 128 * RBFP, GP, HDIM, E, HDIM, RBFP, nullptr, 0, stream); // recompute pre-gate
        k_silu_f32<<<cdiv(EH, B256), B256, 0, stream>>>(GP, GATE, EH);
        k_zero<<<cdiv(EH, B256), B256, 0, stream>>>(GG, EH);
        k_msg_bwd<<<cdiv((long long)LE * 32, B256), B256, 0, stream>>>(AGG, GATE, X[l], nei, GX, GG, LE);
        k_dsilu_bf16<<<cdiv(EH, B256), B256, 0, stream>>>(GG, GP, b_x, EH);
        gemm(b_x, HDIM, bWrT + (size_t)l * 128 * HDIM, g_grbf, RBFD, E, RBFD, HDIM, nullptr, 1, stream);
    }

    k_dsilu_bf16<<<cdiv(EH, B256), B256, 0, stream>>>(GX, SV, b_x, EH);
    gemm(b_x, HDIM, bWvT, GRSH, SHDIM, E, SHDIM, HDIM, nullptr, 0, stream);          // g_rsh
    k_f32_to_bf16<<<cdiv(EH, B256), B256, 0, stream>>>(GX, b_x, EH);
    gemm(b_x, HDIM, bW_embT, Y[0], HDIM, E, HDIM, HDIM, nullptr, 0, stream);         // g_act1 (reuse Y0)
    k_dsilu_bf16<<<cdiv(EH, B256), B256, 0, stream>>>(Y[0], P1, b_x, EH);
    gemm(b_x, HDIM, bW_matT, GEA, FDIM, E, FDIM, HDIM, nullptr, 0, stream);          // grad wrt (eattr*env)

    k_zero<<<cdiv((long long)NN * 3, B256), B256, 0, stream>>>(FACC, (long long)NN * 3);
    k_edge_bwd<<<cdiv(E, B256), B256, 0, stream>>>(edge_attr, GEA, GRSH, g_grbf, g_envg,
                                                   g_ux, g_uy, g_uz, g_dd, g_env, g_denv,
                                                   eidx, FACC, E);
    k_einsum<<<cdiv((long long)E * 3, B256), B256, 0, stream>>>(eag, GEA, eidx, FACC, E);
    k_force_out<<<cdiv((long long)NN * 3, B256), B256, 0, stream>>>(FACC, out_f + NGRAPH,
                                                                    (long long)NN * 3);
    (void)n_in; (void)out_size; (void)ws_size;
}
